// PointNetSetAbstraction_32512902431504
// MI455X (gfx1250) — compile-verified, matrix operand order checked by reading, not run on hardware
//
#include <hip/hip_runtime.h>

// ---------------------------------------------------------------------------
// PointNet++ Set Abstraction, fused for MI455X (gfx1250, wave32, WMMA)
//   outputs: new_xyz (8,2048,3) ++ new_points (8,2048,128), f32
// ---------------------------------------------------------------------------

typedef __attribute__((ext_vector_type(16))) _Float16 v16h;
typedef __attribute__((ext_vector_type(8)))  _Float16 v8h;
typedef __attribute__((ext_vector_type(8)))  float    v8f;

#define B_   8
#define N_   8192
#define S_   2048
#define K_   32
#define R2_  0.0625f     // RADIUS^2
#define ROWSTRIDE 64     // halfs per activation row in LDS

// ===========================================================================
// 1) Weight packing: fold BN into conv, emit f16 B-operand fragments laid out
//    exactly per the 16-bit 32x16 WMMA operand layout (lane = N & K-half,
//    16 contiguous f16 per lane) so the GEMM loop does pure b128 loads.
// ===========================================================================
__global__ void pack_kernel(const float* __restrict__ W, const float* __restrict__ bconv,
                            const float* __restrict__ g, const float* __restrict__ bt,
                            const float* __restrict__ m, const float* __restrict__ var,
                            int Cin, int Cout, int KB, int NT,
                            _Float16* __restrict__ frag, float* __restrict__ biasOut)
{
    int idx = blockIdx.x * blockDim.x + threadIdx.x;
    if (idx < Cout) {
        float sc = g[idx] * rsqrtf(var[idx] + 1e-5f);
        biasOut[idx] = sc * (bconv[idx] - m[idx]) + bt[idx];
    }
    int total = KB * NT * 512;                 // 32 lanes * 16 halfs per (kb,nt)
    if (idx >= total) return;
    int e    = idx & 15;
    int lane = (idx >> 4) & 31;
    int t    = idx >> 9;
    int nt   = t % NT;
    int kb   = t / NT;
    int v = e >> 1, h = e & 1, half = lane >> 4;
    int n = nt * 16 + (lane & 15);
    int K = kb * 32 + (v < 4 ? (8 * half + 2 * v + h)
                             : (16 + 8 * half + 2 * (v - 4) + h));
    float val = 0.f;
    if (K < Cin) {
        float sc = g[n] * rsqrtf(var[n] + 1e-5f);
        val = sc * W[n * Cin + K];
    }
    frag[idx] = (_Float16)val;
}

// ===========================================================================
// 2) Farthest point sampling: 1 block / batch, 1024 threads, dist state in
//    registers, two-stage argmax (wave shfl + LDS). Also emits new_xyz.
// ===========================================================================
__global__ void fps_kernel(const float* __restrict__ xyz, float* __restrict__ new_xyz)
{
    const int b   = blockIdx.x;
    const int tid = threadIdx.x;           // 0..1023
    const int wv  = tid >> 5;
    const int ln  = tid & 31;
    const float* base = xyz + (size_t)b * N_ * 3;

    float dist[8];
#pragma unroll
    for (int j = 0; j < 8; ++j) dist[j] = 1e10f;

    __shared__ float wmax[32];
    __shared__ int   wmaxi[32];
    __shared__ int   farSlot;

    int far = 0;
    for (int it = 0; it < S_; ++it) {
        if (tid == 0) {
            float* o = new_xyz + ((size_t)b * S_ + it) * 3;
            o[0] = base[3 * far + 0];
            o[1] = base[3 * far + 1];
            o[2] = base[3 * far + 2];
        }
        const float cx = base[3 * far + 0];
        const float cy = base[3 * far + 1];
        const float cz = base[3 * far + 2];

        float m = -1.f; int mi = 0;
#pragma unroll
        for (int j = 0; j < 8; ++j) {
            int i = j * 1024 + tid;
            float dx = base[3 * i + 0] - cx;
            float dy = base[3 * i + 1] - cy;
            float dz = base[3 * i + 2] - cz;
            float d  = dx * dx + dy * dy + dz * dz;
            float nd = fminf(dist[j], d);
            dist[j] = nd;
            if (nd > m) { m = nd; mi = i; }      // smallest i among maxima
        }
        // wave32 argmax (tie -> smaller index, matching jnp.argmax)
#pragma unroll
        for (int off = 16; off > 0; off >>= 1) {
            float om = __shfl_xor(m, off, 32);
            int   oi = __shfl_xor(mi, off, 32);
            if (om > m || (om == m && oi < mi)) { m = om; mi = oi; }
        }
        if (ln == 0) { wmax[wv] = m; wmaxi[wv] = mi; }
        __syncthreads();
        if (wv == 0) {
            float m2 = wmax[ln]; int mi2 = wmaxi[ln];
#pragma unroll
            for (int off = 16; off > 0; off >>= 1) {
                float om = __shfl_xor(m2, off, 32);
                int   oi = __shfl_xor(mi2, off, 32);
                if (om > m2 || (om == m2 && oi < mi2)) { m2 = om; mi2 = oi; }
            }
            if (ln == 0) farSlot = mi2;
        }
        __syncthreads();
        far = farSlot;
    }
}

// ===========================================================================
// 3) Ball query: exact 32-NN per query (one wave each): per-lane sorted
//    top-32 over a 256-point strip, then 32-step wave-wide min-merge.
//    Neighbors with d2 > R^2 are replaced by the nearest (reference semantics).
// ===========================================================================
__global__ void ballq_kernel(const float* __restrict__ xyz,
                             const float* __restrict__ new_xyz,
                             int* __restrict__ ball_idx)
{
    const int tid  = threadIdx.x;
    const int wv   = tid >> 5;
    const int lane = tid & 31;
    const int q    = blockIdx.x * (blockDim.x >> 5) + wv;   // 0..B*S-1
    const int b    = q >> 11;

    const float* nz = new_xyz + (size_t)q * 3;
    const float cx = nz[0], cy = nz[1], cz = nz[2];
    const float* base = xyz + (size_t)b * N_ * 3;

    float bd[32];
    int   bi[32];
#pragma unroll
    for (int j = 0; j < 32; ++j) { bd[j] = 3.4e38f; bi[j] = 0; }

    for (int i = lane; i < N_; i += 32) {
        float dx = base[3 * i + 0] - cx;
        float dy = base[3 * i + 1] - cy;
        float dz = base[3 * i + 2] - cz;
        float d  = dx * dx + dy * dy + dz * dz;
        if (d < bd[31]) {
            int p = 31;
            while (p > 0 && bd[p - 1] > d) {
                bd[p] = bd[p - 1]; bi[p] = bi[p - 1]; --p;
            }
            bd[p] = d; bi[p] = i;
        }
    }

    int ptr = 0;
    int nearest = 0;
    for (int k = 0; k < K_; ++k) {
        float d  = bd[ptr];
        int   ii = bi[ptr];
        int   wl = lane;
#pragma unroll
        for (int off = 16; off > 0; off >>= 1) {
            float od = __shfl_xor(d, off, 32);
            int   oi = __shfl_xor(ii, off, 32);
            int   ol = __shfl_xor(wl, off, 32);
            if (od < d || (od == d && ol < wl)) { d = od; ii = oi; wl = ol; }
        }
        if (lane == wl) ++ptr;
        if (k == 0) nearest = ii;
        int sel = (k == 0 || d <= R2_) ? ii : nearest;
        if (lane == 0) ball_idx[(q << 5) + k] = sel;
    }
}

// ===========================================================================
// 4) Fused group + 3-layer MLP (WMMA f16->f32) + max-pool.
//    One wave per query point (32 samples = 2 M-tiles), 4 waves / block.
//    A fragments loaded once per M-tile and reused across all N-tiles.
// ===========================================================================
__device__ __forceinline__ v16h load_afrag(const _Float16* buf, int mt, int kb, int lane)
{
    const int half = lane >> 4, m = lane & 15;
    const _Float16* r = buf + (mt * 16 + m) * ROWSTRIDE + kb * 32;
    v8h lo = *(const v8h*)(r + 8 * half);
    v8h hi = *(const v8h*)(r + 16 + 8 * half);
    v16h a;
#pragma unroll
    for (int i = 0; i < 8; ++i) { a[i] = lo[i]; a[i + 8] = hi[i]; }
    return a;
}

__device__ __forceinline__ v16h load_bfrag(const _Float16* frags, int kb, int nt, int NT, int lane)
{
    const _Float16* p = frags + ((size_t)((kb * NT + nt) * 32 + lane)) * 16;
    v8h lo = *(const v8h*)p;
    v8h hi = *(const v8h*)(p + 8);
    v16h bm;
#pragma unroll
    for (int i = 0; i < 8; ++i) { bm[i] = lo[i]; bm[i + 8] = hi[i]; }
    return bm;
}

template<int KB, int NT>
__device__ __forceinline__ void mlp_layer(const _Float16* bufIn, _Float16* bufOut,
                                          const _Float16* frag, const float* bias,
                                          int lane)
{
    const int half = lane >> 4, n = lane & 15;
#pragma unroll
    for (int mt = 0; mt < 2; ++mt) {
        v16h a[KB];
#pragma unroll
        for (int kb = 0; kb < KB; ++kb) a[kb] = load_afrag(bufIn, mt, kb, lane);
#pragma unroll
        for (int nt = 0; nt < NT; ++nt) {
            v8f acc = {};
#pragma unroll
            for (int kb = 0; kb < KB; ++kb) {
                v16h bm = load_bfrag(frag, kb, nt, NT, lane);
                acc = __builtin_amdgcn_wmma_f32_16x16x32_f16(
                        false, a[kb], false, bm, (short)0, acc, false, false);
            }
            float bv = bias[nt * 16 + n];
            _Float16* col = bufOut + nt * 16 + n + (mt * 16 + half * 8) * ROWSTRIDE;
#pragma unroll
            for (int r = 0; r < 8; ++r) {
                float v = acc[r] + bv;
                v = v > 0.f ? v : 0.f;
                col[r * ROWSTRIDE] = (_Float16)v;
            }
        }
    }
}

__global__ void __launch_bounds__(128)
mlp_kernel(const float* __restrict__ xyz, const float* __restrict__ points,
           const int* __restrict__ ball_idx, const float* __restrict__ new_xyz,
           const _Float16* __restrict__ frag0, const float* __restrict__ bias0,
           const _Float16* __restrict__ frag1, const float* __restrict__ bias1,
           const _Float16* __restrict__ frag2, const float* __restrict__ bias2,
           float* __restrict__ out_points)
{
    __shared__ __align__(16) _Float16 smem[4 * 2 * 32 * ROWSTRIDE];  // 32 KB

    const int tid  = threadIdx.x;
    const int wv   = tid >> 5;
    const int lane = tid & 31;
    const int q    = blockIdx.x * 4 + wv;       // query id, 0..B*S-1
    const int b    = q >> 11;

    _Float16* bufA = smem + wv * 2 * 32 * ROWSTRIDE;
    _Float16* bufB = bufA + 32 * ROWSTRIDE;

    // ---- build X0: row = sample k (this lane), cols 0..8 features, 9..31 zero
    {
        int idx = ball_idx[(q << 5) + lane];
        const float* p3 = xyz    + ((size_t)b * N_ + idx) * 3;
        const float* p6 = points + ((size_t)b * N_ + idx) * 6;
        const float* nz = new_xyz + (size_t)q * 3;
        _Float16 f[32];
        f[0] = (_Float16)(p3[0] - nz[0]);
        f[1] = (_Float16)(p3[1] - nz[1]);
        f[2] = (_Float16)(p3[2] - nz[2]);
#pragma unroll
        for (int c = 0; c < 6; ++c) f[3 + c] = (_Float16)p6[c];
#pragma unroll
        for (int c = 9; c < 32; ++c) f[c] = (_Float16)0.f;
        _Float16* row = bufA + lane * ROWSTRIDE;
#pragma unroll
        for (int c = 0; c < 4; ++c)
            *(v8h*)(row + 8 * c) = *(const v8h*)(f + 8 * c);
    }
    __syncthreads();

    // layer0: 9(->32 padded) -> 64     (bufA -> bufB)
    mlp_layer<1, 4>(bufA, bufB, frag0, bias0, lane);
    __syncthreads();
    // layer1: 64 -> 64                 (bufB -> bufA)
    mlp_layer<2, 4>(bufB, bufA, frag1, bias1, lane);
    __syncthreads();

    // layer2: 64 -> 128, fused max over K=32 rows
    {
        const int n = lane & 15;
        float* outp = out_points + (size_t)q * 128;
        v16h a0[2], a1[2];
#pragma unroll
        for (int kb = 0; kb < 2; ++kb) {
            a0[kb] = load_afrag(bufA, 0, kb, lane);
            a1[kb] = load_afrag(bufA, 1, kb, lane);
        }
#pragma unroll
        for (int nt = 0; nt < 8; ++nt) {
            float bv = bias2[nt * 16 + n];
            v8f acc0 = {}, acc1 = {};
#pragma unroll
            for (int kb = 0; kb < 2; ++kb) {
                v16h bm = load_bfrag(frag2, kb, nt, 8, lane);
                acc0 = __builtin_amdgcn_wmma_f32_16x16x32_f16(
                        false, a0[kb], false, bm, (short)0, acc0, false, false);
                acc1 = __builtin_amdgcn_wmma_f32_16x16x32_f16(
                        false, a1[kb], false, bm, (short)0, acc1, false, false);
            }
            float mval = 0.f;   // post-ReLU values are >= 0
#pragma unroll
            for (int r = 0; r < 8; ++r) {
                mval = fmaxf(mval, acc0[r] + bv);
                mval = fmaxf(mval, acc1[r] + bv);
            }
            float o = __shfl_xor(mval, 16, 32);   // combine M halves
            mval = fmaxf(mval, o);
            if (lane < 16) outp[nt * 16 + n] = mval;
        }
    }
}

// ===========================================================================
// launch
// ===========================================================================
extern "C" void kernel_launch(void* const* d_in, const int* in_sizes, int n_in,
                              void* d_out, int out_size, void* d_ws, size_t ws_size,
                              hipStream_t stream)
{
    const float* xyz    = (const float*)d_in[0];
    const float* points = (const float*)d_in[1];
    const float* w0 = (const float*)d_in[2],  *b0 = (const float*)d_in[3];
    const float* g0 = (const float*)d_in[4],  *bt0 = (const float*)d_in[5];
    const float* m0 = (const float*)d_in[6],  *v0 = (const float*)d_in[7];
    const float* w1 = (const float*)d_in[8],  *b1 = (const float*)d_in[9];
    const float* g1 = (const float*)d_in[10], *bt1 = (const float*)d_in[11];
    const float* m1 = (const float*)d_in[12], *v1 = (const float*)d_in[13];
    const float* w2 = (const float*)d_in[14], *b2 = (const float*)d_in[15];
    const float* g2 = (const float*)d_in[16], *bt2 = (const float*)d_in[17];
    const float* m2 = (const float*)d_in[18], *v2 = (const float*)d_in[19];

    char* ws = (char*)d_ws;
    size_t off = 0;
    int* ball_idx = (int*)(ws + off);        off += (size_t)B_ * S_ * K_ * 4;  // 2 MB
    _Float16* frag0 = (_Float16*)(ws + off); off += 1 * 4 * 512 * 2;           // 4 KB
    _Float16* frag1 = (_Float16*)(ws + off); off += 2 * 4 * 512 * 2;           // 8 KB
    _Float16* frag2 = (_Float16*)(ws + off); off += 2 * 8 * 512 * 2;           // 16 KB
    float* bias0 = (float*)(ws + off);       off += 64 * 4;
    float* bias1 = (float*)(ws + off);       off += 64 * 4;
    float* bias2 = (float*)(ws + off);       off += 128 * 4;

    float* out        = (float*)d_out;
    float* new_xyz    = out;                       // (B,S,3)
    float* new_points = out + (size_t)B_ * S_ * 3; // (B,S,128)

    // fold BN + pack weight fragments
    pack_kernel<<<8,  256, 0, stream>>>(w0, b0, g0, bt0, m0, v0,  9,  64, 1, 4, frag0, bias0);
    pack_kernel<<<16, 256, 0, stream>>>(w1, b1, g1, bt1, m1, v1, 64,  64, 2, 4, frag1, bias1);
    pack_kernel<<<32, 256, 0, stream>>>(w2, b2, g2, bt2, m2, v2, 64, 128, 2, 8, frag2, bias2);

    // FPS (also writes new_xyz)
    fps_kernel<<<B_, 1024, 0, stream>>>(xyz, new_xyz);

    // ball query: 16384 queries, 8 waves / block
    ballq_kernel<<<(B_ * S_) / 8, 256, 0, stream>>>(xyz, new_xyz, ball_idx);

    // fused group + MLP + maxpool: 4 queries / block
    mlp_kernel<<<(B_ * S_) / 4, 128, 0, stream>>>(xyz, points, ball_idx, new_xyz,
                                                  frag0, bias0, frag1, bias1, frag2, bias2,
                                                  new_points);
}
